// Raindrop_15985868276153
// MI455X (gfx1250) — matrix operand order repaired and checked by reading, not compiled, not run on hardware
//
#include <hip/hip_runtime.h>
#include <hip/hip_bf16.h>
#include <math.h>

typedef __attribute__((ext_vector_type(16))) _Float16 v16h;
typedef __attribute__((ext_vector_type(8)))  float    v8f;

#define T_LEN   215
#define BATCH   256
#define MROWS   (T_LEN * BATCH)   // 55040 (multiple of 16)
#define D_T     72
#define NHEAD   4
#define DHEAD   18
#define NNODES  36
#define TPAD    224               // 14 * 16

// ---------------------------------------------------------------------------
// WMMA helpers (contiguous k order, k = 16*(lane/16) + e, wave32)
// ---------------------------------------------------------------------------
__device__ __forceinline__ int laneid() { return threadIdx.x & 31; }

__device__ __forceinline__ v8f wmma_f16(v16h a, v16h b, v8f c) {
  return __builtin_amdgcn_wmma_f32_16x16x32_f16(false, a, false, b, (short)0, c,
                                                false, false);
}

// A tile: 16x32 f16, row-major source with leading dim ld.
// lane<16 : row=lane,    K = {0..15}
// lane>=16: row=lane-16, K = {16..31}
__device__ __forceinline__ v16h load_A16(const _Float16* p, int ld) {
  int lane = laneid();
  const _Float16* row = p + (size_t)(lane & 15) * ld;
  int kb = (lane < 16) ? 0 : 16;
  v16h a;
#pragma unroll
  for (int e = 0; e < 16; ++e) a[e] = row[kb + e];
  return a;
}

// B tile for A @ W^T, W row-major [N,K] with leading dim ld.
// B[k][n] = W[n][k]; lane n<16 -> b[e]=W[n][e]; lane>=16 -> b[e]=W[n-16][16+e]
// => 16 contiguous halves per lane.
__device__ __forceinline__ v16h load_Bt16(const _Float16* p, int ld) {
  int lane = laneid();
  const _Float16* row = p + (size_t)(lane & 15) * ld + ((lane < 16) ? 0 : 16);
  v16h b;
#pragma unroll
  for (int e = 0; e < 16; ++e) b[e] = row[e];
  return b;
}

// B tile for A @ B, B row-major [K,N] with leading dim ld.
__device__ __forceinline__ v16h load_Bn16(const _Float16* p, int ld) {
  int lane = laneid();
  int n = lane & 15;
  int kb = (lane < 16) ? 0 : 16;
  v16h b;
#pragma unroll
  for (int e = 0; e < 16; ++e) b[e] = p[(size_t)(kb + e) * ld + n];
  return b;
}

// ---------------------------------------------------------------------------
// f32 -> padded f16 conversion (zero-fills row/col padding)
// ---------------------------------------------------------------------------
__global__ void cvt_pad(const float* __restrict__ in, int ldin, int rows, int cols,
                        _Float16* __restrict__ out, int ldo, int rowsPad) {
  size_t total = (size_t)rowsPad * ldo;
  for (size_t i = (size_t)blockIdx.x * blockDim.x + threadIdx.x; i < total;
       i += (size_t)gridDim.x * blockDim.x) {
    int r = (int)(i / ldo), c = (int)(i % ldo);
    float v = (r < rows && c < cols) ? in[(size_t)r * ldin + c] : 0.f;
    out[i] = (_Float16)v;
  }
}

// ---------------------------------------------------------------------------
// Generic WMMA GEMM: out = act((A @ W^T + bias) * scale) + residual
// A: f16 [M, Kpad] lda; W: f16 [Npad, Kpad] ldw (row-major, padded w/ zeros)
// One wave per 16x16 C tile, chained 16x16x32 WMMAs over kblk K-blocks.
// ---------------------------------------------------------------------------
__global__ __launch_bounds__(32)
void gemm16(const _Float16* __restrict__ A, int lda,
            const _Float16* __restrict__ W, int ldw,
            const float* __restrict__ bias, int biasMax,
            const float* __restrict__ res, int ldres,
            float* __restrict__ Cf, int ldcf,
            _Float16* __restrict__ Ch, int ldch,
            int M, int N, int kblk, float scale, int relu) {
  int lane = laneid();
  int n0 = blockIdx.x * 16;
  int m0 = blockIdx.y * 16;
  int col = n0 + (lane & 15);
  float bv = 0.f;
  if (bias) bv = bias[col < biasMax ? col : (biasMax - 1)];
  v8f c;
#pragma unroll
  for (int r = 0; r < 8; ++r) c[r] = bv;
  const _Float16* ap = A + (size_t)m0 * lda;
  const _Float16* wp = W + (size_t)n0 * ldw;
  for (int kb = 0; kb < kblk; ++kb) {
    v16h a = load_A16(ap + kb * 32, lda);
    v16h b = load_Bt16(wp + kb * 32, ldw);
    c = wmma_f16(a, b, c);
  }
  if (col < N) {
    int rbase = (lane < 16) ? 0 : 8;
#pragma unroll
    for (int r = 0; r < 8; ++r) {
      int mm = m0 + rbase + r;
      if (mm >= M) break;
      float v = c[r] * scale;
      if (relu) v = fmaxf(v, 0.f);
      if (res) v += res[(size_t)mm * ldres + col];
      if (Cf) Cf[(size_t)mm * ldcf + col] = v;
      if (Ch) Ch[(size_t)mm * ldch + col] = (_Float16)v;
    }
  }
}

// ---------------------------------------------------------------------------
// Graph attention over 36 nodes (one block per batch element).
// Computes A=softmax(q k^T / 6), adds msg=A v into gskip, dumps A flat f16.
// ---------------------------------------------------------------------------
__global__ __launch_bounds__(128)
void graph_attn(const float* __restrict__ qg, const float* __restrict__ kg,
                const float* __restrict__ vg, float* __restrict__ gskip,
                _Float16* __restrict__ aflat) {
  __shared__ float qs[NNODES * NNODES], ks[NNODES * NNODES];
  __shared__ float vs[NNODES * NNODES], sc[NNODES * NNODES];
  int b = blockIdx.x, tid = threadIdx.x;
  for (int e = tid; e < 1296; e += 128) {
    int t = e / 36, d = e % 36;
    size_t ro = ((size_t)t * BATCH + b) * 36 + d;
    qs[e] = qg[ro]; ks[e] = kg[ro]; vs[e] = vg[ro];
  }
  __syncthreads();
  for (int e = tid; e < 1296; e += 128) {
    int i = e / 36, j = e % 36;
    float s = 0.f;
    for (int d = 0; d < 36; ++d) s += qs[i * 36 + d] * ks[j * 36 + d];
    sc[e] = s * (1.0f / 6.0f);
  }
  __syncthreads();
  if (tid < 36) {
    float m = -1e30f;
    for (int j = 0; j < 36; ++j) m = fmaxf(m, sc[tid * 36 + j]);
    float s = 0.f;
    for (int j = 0; j < 36; ++j) {
      float e2 = expf(sc[tid * 36 + j] - m);
      sc[tid * 36 + j] = e2; s += e2;
    }
    float inv = 1.f / s;
    for (int j = 0; j < 36; ++j) sc[tid * 36 + j] *= inv;
  }
  __syncthreads();
  for (int e = tid; e < 1296; e += 128) {
    aflat[(size_t)b * 1312 + e] = (_Float16)sc[e];
    int t = e / 36, d = e % 36;
    float s = 0.f;
    for (int j = 0; j < 36; ++j) s += sc[t * 36 + j] * vs[j * 36 + d];
    gskip[((size_t)t * BATCH + b) * 36 + d] += s;
  }
}

// ---------------------------------------------------------------------------
// distance = mean over 256x256 of sqrt(max(sq_i+sq_j-2G_ij,0)) where >0
// ---------------------------------------------------------------------------
__global__ __launch_bounds__(256)
void dist_reduce(const float* __restrict__ G, float* __restrict__ out) {
  __shared__ float red[256];
  float acc = 0.f;
  for (int idx = threadIdx.x; idx < 65536; idx += 256) {
    int i = idx >> 8, j = idx & 255;
    float sqi = G[i * 256 + i], sqj = G[j * 256 + j];
    float d2 = fmaxf(sqi + sqj - 2.f * G[i * 256 + j], 0.f);
    if (d2 > 0.f) acc += sqrtf(d2);
  }
  red[threadIdx.x] = acc;
  __syncthreads();
  for (int o = 128; o; o >>= 1) {
    if (threadIdx.x < o) red[threadIdx.x] += red[threadIdx.x + o];
    __syncthreads();
  }
  if (threadIdx.x == 0) out[0] = red[0] * (1.0f / 65536.0f);
}

// ---------------------------------------------------------------------------
// h = concat(gout, pe); writes f32 h and padded f16 h16
// ---------------------------------------------------------------------------
__global__ void build_h(const float* __restrict__ gskip, const float* __restrict__ times,
                        float* __restrict__ h, _Float16* __restrict__ h16) {
  size_t i = (size_t)blockIdx.x * blockDim.x + threadIdx.x;
  if (i >= (size_t)MROWS * 72) return;
  int row = (int)(i / 72), d = (int)(i % 72);
  float v;
  if (d < 36) {
    v = gskip[(size_t)row * 36 + d];
  } else {
    int dd = d - 36;
    int t = row / BATCH, b = row % BATCH;
    float ts = powf(215.0f, (float)(dd % 18) * (1.0f / 17.0f));
    float arg = times[(size_t)t * BATCH + b] / ts;
    v = (dd < 18) ? sinf(arg) : cosf(arg);
  }
  h[(size_t)row * 72 + d] = v;
  h16[(size_t)row * 96 + d] = (_Float16)v;
}

// ---------------------------------------------------------------------------
// Masked multi-head attention: one wave per (q-tile, head, batch).
// Two-pass: full 16x224 score strip in LDS, softmax, then P @ V via WMMA.
// qkv16: f16 [T*B, 224] (q|k|v each 72 wide). ctx16: f16 [T*B, 96].
// ---------------------------------------------------------------------------
__global__ __launch_bounds__(32)
void attention(const _Float16* __restrict__ qkv, const int* __restrict__ lengths,
               _Float16* __restrict__ ctx16) {
  __shared__ _Float16 Qs[16 * 32];
  __shared__ _Float16 KVs[32 * 32];
  __shared__ float    Ss[16 * TPAD];
  __shared__ _Float16 Ps[16 * TPAD];
  int lane = threadIdx.x;
  int qt = blockIdx.x, hh = blockIdx.y, b = blockIdx.z;
  int tq0 = qt * 16;
  int len = lengths[b];

  for (int idx = lane; idx < 16 * 32; idx += 32) {
    int r = idx >> 5, d = idx & 31;
    int t = tq0 + r;
    _Float16 v = (_Float16)0.f;
    if (t < T_LEN && d < DHEAD)
      v = qkv[((size_t)t * BATCH + b) * 224 + hh * DHEAD + d];
    Qs[idx] = v;
  }
  __syncthreads();
  v16h aq = load_A16(Qs, 32);

  // pass 1: scores
  for (int kb = 0; kb < 7; ++kb) {
    for (int idx = lane; idx < 32 * 32; idx += 32) {
      int r = idx >> 5, d = idx & 31;
      int t = kb * 32 + r;
      _Float16 v = (_Float16)0.f;
      if (t < T_LEN && d < DHEAD)
        v = qkv[((size_t)t * BATCH + b) * 224 + 72 + hh * DHEAD + d];
      KVs[idx] = v;
    }
    __syncthreads();
    v8f c0, c1;
#pragma unroll
    for (int r = 0; r < 8; ++r) { c0[r] = 0.f; c1[r] = 0.f; }
    c0 = wmma_f16(aq, load_Bt16(KVs, 32), c0);
    c1 = wmma_f16(aq, load_Bt16(KVs + 16 * 32, 32), c1);
    {
      int n = lane & 15, rb = (lane < 16) ? 0 : 8;
#pragma unroll
      for (int r = 0; r < 8; ++r) {
        Ss[(rb + r) * TPAD + kb * 32 + n] = c0[r];
        Ss[(rb + r) * TPAD + kb * 32 + 16 + n] = c1[r];
      }
    }
    __syncthreads();
  }

  // softmax: 2 lanes per row, 112 cols each
  {
    int r = lane & 15, half = lane >> 4;
    const float rsq = 0.23570226039551584f;  // 1/sqrt(18)
    int c0i = half * 112;
    float m = -1e30f;
    for (int c2 = c0i; c2 < c0i + 112; ++c2) {
      float s = Ss[r * TPAD + c2];
      s = (c2 < len && c2 < T_LEN) ? s * rsq : -1e30f;
      Ss[r * TPAD + c2] = s;
      m = fmaxf(m, s);
    }
    m = fmaxf(m, __shfl_xor(m, 16, 32));
    float sum = 0.f;
    for (int c2 = c0i; c2 < c0i + 112; ++c2) {
      float e = expf(Ss[r * TPAD + c2] - m);
      Ss[r * TPAD + c2] = e;
      sum += e;
    }
    sum += __shfl_xor(sum, 16, 32);
    float inv = 1.f / sum;
    for (int c2 = c0i; c2 < c0i + 112; ++c2)
      Ps[r * TPAD + c2] = (_Float16)(Ss[r * TPAD + c2] * inv);
  }
  __syncthreads();

  // pass 2: ctx = P @ V
  v8f o0, o1;
#pragma unroll
  for (int r = 0; r < 8; ++r) { o0[r] = 0.f; o1[r] = 0.f; }
  for (int kb = 0; kb < 7; ++kb) {
    for (int idx = lane; idx < 32 * 32; idx += 32) {
      int r = idx >> 5, d = idx & 31;
      int t = kb * 32 + r;
      _Float16 v = (_Float16)0.f;
      if (t < T_LEN && d < DHEAD)
        v = qkv[((size_t)t * BATCH + b) * 224 + 144 + hh * DHEAD + d];
      KVs[idx] = v;
    }
    __syncthreads();
    v16h ap = load_A16(Ps + kb * 32, TPAD);
    o0 = wmma_f16(ap, load_Bn16(KVs, 32), o0);
    o1 = wmma_f16(ap, load_Bn16(KVs + 16, 32), o1);
    __syncthreads();
  }
  {
    int n = lane & 15, rb = (lane < 16) ? 0 : 8;
#pragma unroll
    for (int r = 0; r < 8; ++r) {
      int t = tq0 + rb + r;
      if (t < T_LEN) {
        size_t base = ((size_t)t * BATCH + b) * 96 + hh * DHEAD;
        ctx16[base + n] = (_Float16)o0[r];
        if (n < 2) ctx16[base + 16 + n] = (_Float16)o1[r];
      }
    }
  }
}

// ---------------------------------------------------------------------------
// LayerNorm over 72 elems; one wave per row, 8 rows per block.
// ---------------------------------------------------------------------------
__global__ __launch_bounds__(256)
void layernorm8(const float* __restrict__ xin, const float* __restrict__ g,
                const float* __restrict__ bb, float* __restrict__ hout,
                _Float16* __restrict__ h16) {
  int row = blockIdx.x * 8 + (threadIdx.x >> 5);
  int lane = threadIdx.x & 31;
  const float* xr = xin + (size_t)row * 72;
  float x0 = xr[lane], x1 = xr[lane + 32];
  float x2 = (lane < 8) ? xr[lane + 64] : 0.f;
  float s1 = x0 + x1 + x2, s2 = x0 * x0 + x1 * x1 + x2 * x2;
  for (int o = 16; o; o >>= 1) {
    s1 += __shfl_xor(s1, o, 32);
    s2 += __shfl_xor(s2, o, 32);
  }
  float mu = s1 * (1.f / 72.f);
  float var = s2 * (1.f / 72.f) - mu * mu;
  float rstd = rsqrtf(var + 1e-5f);
  float* ho = hout + (size_t)row * 72;
  _Float16* hh = h16 + (size_t)row * 96;
  {
    float y = (x0 - mu) * rstd * g[lane] + bb[lane];
    ho[lane] = y; hh[lane] = (_Float16)y;
  }
  {
    int d = lane + 32;
    float y = (x1 - mu) * rstd * g[d] + bb[d];
    ho[d] = y; hh[d] = (_Float16)y;
  }
  if (lane < 8) {
    int d = lane + 64;
    float y = (x2 - mu) * rstd * g[d] + bb[d];
    ho[d] = y; hh[d] = (_Float16)y;
  }
}

// ---------------------------------------------------------------------------
// pooled (masked mean, /(len+1)) + static embedding -> feat [B,108]
// ---------------------------------------------------------------------------
__global__ __launch_bounds__(128)
void pool_feat(const float* __restrict__ h, const int* __restrict__ lengths,
               const float* __restrict__ statc, const float* __restrict__ Wemb,
               const float* __restrict__ bemb, float* __restrict__ feat) {
  int b = blockIdx.x, d = threadIdx.x;
  int len = lengths[b];
  if (d < 72) {
    float s = 0.f;
    int lim = len < T_LEN ? len : T_LEN;
    for (int t = 0; t < lim; ++t) s += h[((size_t)t * BATCH + b) * 72 + d];
    feat[(size_t)b * 108 + d] = s / ((float)len + 1.f);
  } else if (d < 108) {
    int dd = d - 72;
    float s = bemb[dd];
    for (int k = 0; k < 9; ++k) s += statc[(size_t)b * 9 + k] * Wemb[(size_t)dd * 9 + k];
    feat[(size_t)b * 108 + d] = s;
  }
}

__global__ __launch_bounds__(128)
void mlp_head(const float* __restrict__ feat, const float* __restrict__ w1,
              const float* __restrict__ b1, const float* __restrict__ w2,
              const float* __restrict__ b2, float* __restrict__ out) {
  __shared__ float hid[108];
  int b = blockIdx.x, j = threadIdx.x;
  if (j < 108) {
    float s = b1[j];
    for (int k = 0; k < 108; ++k) s += feat[(size_t)b * 108 + k] * w1[(size_t)j * 108 + k];
    hid[j] = fmaxf(s, 0.f);
  }
  __syncthreads();
  if (j < 2) {
    float s = b2[j];
    for (int k = 0; k < 108; ++k) s += hid[k] * w2[(size_t)j * 108 + k];
    out[(size_t)b * 2 + j] = s;
  }
}

// ---------------------------------------------------------------------------
extern "C" void kernel_launch(void* const* d_in, const int* in_sizes, int n_in,
                              void* d_out, int out_size, void* d_ws, size_t ws_size,
                              hipStream_t stream) {
  (void)in_sizes; (void)n_in; (void)out_size; (void)ws_size;
  const float* src    = (const float*)d_in[0];
  const float* statc  = (const float*)d_in[1];
  const float* times  = (const float*)d_in[2];
  const int*   lens   = (const int*)d_in[3];
  const float* W_enc  = (const float*)d_in[4];
  const float* b_enc  = (const float*)d_in[5];
  const float* W_emb  = (const float*)d_in[6];
  const float* b_emb  = (const float*)d_in[7];
  const float* Wq     = (const float*)d_in[8];
  const float* bq     = (const float*)d_in[9];
  const float* Wk     = (const float*)d_in[10];
  const float* bk     = (const float*)d_in[11];
  const float* Wv     = (const float*)d_in[12];
  const float* bv     = (const float*)d_in[13];
  const float* Wskip  = (const float*)d_in[14];
  const float* bskip  = (const float*)d_in[15];
  const float* einw   = (const float*)d_in[16];
  const float* einb   = (const float*)d_in[17];
  const float* eoutw  = (const float*)d_in[18];
  const float* eoutb  = (const float*)d_in[19];
  const float* ff1w   = (const float*)d_in[20];
  const float* ff1b   = (const float*)d_in[21];
  const float* ff2w   = (const float*)d_in[22];
  const float* ff2b   = (const float*)d_in[23];
  const float* ln1g   = (const float*)d_in[24];
  const float* ln1b   = (const float*)d_in[25];
  const float* ln2g   = (const float*)d_in[26];
  const float* ln2b   = (const float*)d_in[27];
  const float* mw1    = (const float*)d_in[28];
  const float* mb1    = (const float*)d_in[29];
  const float* mw2    = (const float*)d_in[30];
  const float* mb2    = (const float*)d_in[31];
  float* out = (float*)d_out;

  char* p = (char*)d_ws;
  auto alloc = [&](size_t bytes) -> char* {
    char* r = p; p += (bytes + 255) & ~(size_t)255; return r;
  };
  _Float16* src16   = (_Float16*)alloc((size_t)MROWS * 64 * 2);
  _Float16* x16     = (_Float16*)alloc((size_t)MROWS * 64 * 2);
  _Float16* h16     = (_Float16*)alloc((size_t)MROWS * 96 * 2);
  _Float16* ctx16   = (_Float16*)alloc((size_t)MROWS * 96 * 2);
  _Float16* ff116   = (_Float16*)alloc((size_t)MROWS * 128 * 2);
  _Float16* qkv16   = (_Float16*)alloc((size_t)MROWS * 224 * 2);
  float*    hbuf    = (float*)alloc((size_t)MROWS * 72 * 4);
  float*    h2      = (float*)alloc((size_t)MROWS * 72 * 4);
  float*    gskip   = (float*)alloc((size_t)MROWS * 36 * 4);
  float*    qg      = (float*)alloc((size_t)9216 * 36 * 4);
  float*    kg      = (float*)alloc((size_t)9216 * 36 * 4);
  float*    vg      = (float*)alloc((size_t)9216 * 36 * 4);
  _Float16* aflat   = (_Float16*)alloc((size_t)256 * 1312 * 2);
  float*    G       = (float*)alloc((size_t)256 * 256 * 4);
  float*    feat    = (float*)alloc((size_t)256 * 108 * 4);
  _Float16* wenc16  = (_Float16*)alloc(48 * 64 * 2);
  _Float16* wq16    = (_Float16*)alloc(48 * 64 * 2);
  _Float16* wk16    = (_Float16*)alloc(48 * 64 * 2);
  _Float16* wv16    = (_Float16*)alloc(48 * 64 * 2);
  _Float16* wsk16   = (_Float16*)alloc(48 * 64 * 2);
  _Float16* ein16   = (_Float16*)alloc((size_t)2 * 224 * 96 * 2);
  _Float16* eout16  = (_Float16*)alloc((size_t)2 * 80 * 96 * 2);
  _Float16* f1w16   = (_Float16*)alloc((size_t)2 * 128 * 96 * 2);
  _Float16* f2w16   = (_Float16*)alloc((size_t)2 * 80 * 128 * 2);

  hipMemsetAsync(x16,   0, (size_t)MROWS * 64 * 2, stream);
  hipMemsetAsync(h16,   0, (size_t)MROWS * 96 * 2, stream);
  hipMemsetAsync(ctx16, 0, (size_t)MROWS * 96 * 2, stream);
  hipMemsetAsync(aflat, 0, (size_t)256 * 1312 * 2, stream);

  auto cvt = [&](const float* in, int ldin, int rows, int cols,
                 _Float16* o, int ldo, int rowsPad) {
    size_t total = (size_t)rowsPad * ldo;
    int grid = (int)((total + 255) / 256);
    cvt_pad<<<grid, 256, 0, stream>>>(in, ldin, rows, cols, o, ldo, rowsPad);
  };
  cvt(src, 72, MROWS, 36, src16, 64, MROWS);
  cvt(W_enc, 36, 36, 36, wenc16, 64, 48);
  cvt(Wq, 36, 36, 36, wq16, 64, 48);
  cvt(Wk, 36, 36, 36, wk16, 64, 48);
  cvt(Wv, 36, 36, 36, wv16, 64, 48);
  cvt(Wskip, 36, 36, 36, wsk16, 64, 48);
  for (int l = 0; l < 2; ++l) {
    cvt(einw + (size_t)l * 216 * 72, 72, 216, 72, ein16 + (size_t)l * 224 * 96, 96, 224);
    cvt(eoutw + (size_t)l * 72 * 72, 72, 72, 72, eout16 + (size_t)l * 80 * 96, 96, 80);
    cvt(ff1w + (size_t)l * 128 * 72, 72, 128, 72, f1w16 + (size_t)l * 128 * 96, 96, 128);
    cvt(ff2w + (size_t)l * 72 * 128, 128, 72, 128, f2w16 + (size_t)l * 80 * 128, 128, 80);
  }

  auto gemm = [&](const _Float16* A, int lda, const _Float16* W, int ldw,
                  const float* bias, const float* res, int ldres,
                  float* Cf, int ldcf, _Float16* Ch, int ldch,
                  int M, int N, int kblk, float scale, int relu) {
    dim3 g((N + 15) / 16, M / 16);
    gemm16<<<g, 32, 0, stream>>>(A, lda, W, ldw, bias, N, res, ldres,
                                 Cf, ldcf, Ch, ldch, M, N, kblk, scale, relu);
  };

  // x = (src[:, :36] @ W_enc^T + b_enc) * 6
  gemm(src16, 64, wenc16, 64, b_enc, nullptr, 0, nullptr, 0, x16, 64,
       MROWS, 36, 2, 6.0f, 0);
  // graph projections (q/k/v on first 36 timesteps; skip on all)
  gemm(x16, 64, wq16, 64, bq, nullptr, 0, qg, 36, nullptr, 0, 9216, 36, 2, 1.f, 0);
  gemm(x16, 64, wk16, 64, bk, nullptr, 0, kg, 36, nullptr, 0, 9216, 36, 2, 1.f, 0);
  gemm(x16, 64, wv16, 64, bv, nullptr, 0, vg, 36, nullptr, 0, 9216, 36, 2, 1.f, 0);
  gemm(x16, 64, wsk16, 64, bskip, nullptr, 0, gskip, 36, nullptr, 0, MROWS, 36, 2, 1.f, 0);
  graph_attn<<<256, 128, 0, stream>>>(qg, kg, vg, gskip, aflat);
  // Gram matrix of flattened attention weights (K=1296 pad 1312) + distance
  gemm(aflat, 1312, aflat, 1312, nullptr, nullptr, 0, G, 256, nullptr, 0,
       256, 256, 41, 1.f, 0);
  dist_reduce<<<1, 256, 0, stream>>>(G, out + 512);
  // h = concat(gout, pe)
  build_h<<<((size_t)MROWS * 72 + 255) / 256, 256, 0, stream>>>(gskip, times, hbuf, h16);

  for (int l = 0; l < 2; ++l) {
    gemm(h16, 96, ein16 + (size_t)l * 224 * 96, 96, einb + (size_t)l * 216,
         nullptr, 0, nullptr, 0, qkv16, 224, MROWS, 216, 3, 1.f, 0);
    attention<<<dim3(14, NHEAD, BATCH), 32, 0, stream>>>(qkv16, lens, ctx16);
    gemm(ctx16, 96, eout16 + (size_t)l * 80 * 96, 96, eoutb + (size_t)l * 72,
         hbuf, 72, h2, 72, nullptr, 0, MROWS, 72, 3, 1.f, 0);
    layernorm8<<<MROWS / 8, 256, 0, stream>>>(h2, ln1g + (size_t)l * 72,
                                              ln1b + (size_t)l * 72, hbuf, h16);
    gemm(h16, 96, f1w16 + (size_t)l * 128 * 96, 96, ff1b + (size_t)l * 128,
         nullptr, 0, nullptr, 0, ff116, 128, MROWS, 128, 3, 1.f, 1);
    gemm(ff116, 128, f2w16 + (size_t)l * 80 * 128, 128, ff2b + (size_t)l * 72,
         hbuf, 72, h2, 72, nullptr, 0, MROWS, 72, 4, 1.f, 0);
    layernorm8<<<MROWS / 8, 256, 0, stream>>>(h2, ln2g + (size_t)l * 72,
                                              ln2b + (size_t)l * 72, hbuf, h16);
  }

  pool_feat<<<256, 128, 0, stream>>>(hbuf, lens, statc, W_emb, b_emb, feat);
  mlp_head<<<256, 128, 0, stream>>>(feat, mw1, mb1, mw2, mb2, out);
}
